// DecoderOnlyLM_51178830299351
// MI455X (gfx1250) — matrix operand order repaired and checked by reading, not compiled, not run on hardware
//
#include <hip/hip_runtime.h>
#include <hip/hip_bf16.h>

// ---------------------------------------------------------------------------
// Model constants (match reference)
// ---------------------------------------------------------------------------
#define Vv    32000
#define Dd    1024
#define Hh    16
#define Ll    8
#define Bb    2
#define Ss    1024
#define DH    64
#define INNER 4096
#define Mtok  (Bb * Ss)          // 2048 token rows

typedef __bf16 bf16_t;
typedef bf16_t v16bf __attribute__((ext_vector_type(16)));
typedef bf16_t v8bf  __attribute__((ext_vector_type(8)));
typedef float  v8f   __attribute__((ext_vector_type(8)));
typedef unsigned int u32;
typedef u32 u32x4 __attribute__((ext_vector_type(4)));
typedef int i32x4  __attribute__((ext_vector_type(4)));
typedef int i32x8  __attribute__((ext_vector_type(8)));

union FragBF { v16bf v; v8bf h[2]; };

__device__ __forceinline__ v8f wmma_bf16(v16bf a, v16bf b, v8f c) {
    // D(f32,16x16) = A(bf16,16x32) * B(bf16,32x16) + C
    return __builtin_amdgcn_wmma_f32_16x16x32_bf16(
        false, a, false, b, (short)0, c, false, false);
}

__device__ __forceinline__ v8f zero8() {
    v8f z;
#pragma unroll
    for (int i = 0; i < 8; ++i) z[i] = 0.0f;
    return z;
}

// ---------------------------------------------------------------------------
// Tensor Data Mover: 2-D bf16 tile (tile_w elems x tile_h rows) from global
// row-major [*, row_stride] into LDS at byte offset lds_off.
// D# layout per CDNA5 ISA (count=1, type=2, ds=2B).
// Issue from one wave only; EXEC is ignored by TDM. Tracked by TENSORcnt.
// ---------------------------------------------------------------------------
__device__ __forceinline__ void tdm_load_2d(const bf16_t* gptr, u32 lds_off,
                                            u32 tile_w, u32 tile_h,
                                            u32 row_stride)
{
    unsigned long long ga = (unsigned long long)(uintptr_t)gptr;
    u32x4 g0;
    g0[0] = 1u;                                        // count=1, no gather
    g0[1] = lds_off;                                   // lds_addr (bytes)
    g0[2] = (u32)ga;                                   // global_addr[31:0]
    g0[3] = (u32)((ga >> 32) & 0x01ffffffu) | (2u << 30); // addr[56:32], type=2
    i32x8 g1;
    g1[0] = (int)(1u << 16);                           // wg_mask=0, data_size=2B
    g1[1] = (int)((tile_w & 0xffffu) << 16);           // tensor_dim0 lo16
    g1[2] = (int)(((tile_w >> 16) & 0xffffu) | ((tile_h & 0xffffu) << 16));
    g1[3] = (int)(((tile_h >> 16) & 0xffffu) | ((tile_w & 0xffffu) << 16)); // tile_dim0
    g1[4] = (int)(tile_h & 0xffffu);                   // tile_dim1 (tile_dim2=0)
    g1[5] = (int)row_stride;                           // tensor_dim0_stride lo32
    g1[6] = 0;
    g1[7] = 0;
    i32x4 z4 = {0, 0, 0, 0};
#if defined(__clang_major__) && (__clang_major__ >= 23)
    i32x8 z8 = {0, 0, 0, 0, 0, 0, 0, 0};
    __builtin_amdgcn_tensor_load_to_lds(g0, g1, z4, z4, z8, 0);
#else
    __builtin_amdgcn_tensor_load_to_lds(g0, g1, z4, z4, 0);
#endif
}

// ---------------------------------------------------------------------------
// fp32 -> bf16 weight panel conversion (each weight used by exactly one GEMM,
// so converting once costs the same traffic but halves GEMM weight reads).
// ---------------------------------------------------------------------------
__global__ __launch_bounds__(256) void cvt_bf16_kernel(
    bf16_t* __restrict__ dst, const float* __restrict__ src)
{
    size_t i = ((size_t)blockIdx.x * 256 + threadIdx.x) * 4;
    float4 f = *(const float4*)(src + i);
    dst[i + 0] = (bf16_t)f.x; dst[i + 1] = (bf16_t)f.y;
    dst[i + 2] = (bf16_t)f.z; dst[i + 3] = (bf16_t)f.w;
}

// ---------------------------------------------------------------------------
// Embedding gather: x[m, :] = tok_emb[ids[m], :]
// ---------------------------------------------------------------------------
__global__ __launch_bounds__(256) void embed_kernel(
    float* __restrict__ x, const float* __restrict__ tok,
    const int* __restrict__ ids)
{
    int idx = blockIdx.x * 256 + threadIdx.x;
    int m = idx >> 10;
    int d = idx & 1023;
    x[idx] = tok[(size_t)ids[m] * Dd + d];
}

// ---------------------------------------------------------------------------
// LayerNorm over D=1024, bf16 output (feeds GEMM A operands only).
// ---------------------------------------------------------------------------
__global__ __launch_bounds__(256) void layernorm_kernel(
    bf16_t* __restrict__ out, const float* __restrict__ in,
    const float* __restrict__ g, const float* __restrict__ b)
{
    __shared__ float red[256];
    const int tid = threadIdx.x;
    const float* x = in + (size_t)blockIdx.x * Dd;
    float v[4];
    float s = 0.f;
#pragma unroll
    for (int i = 0; i < 4; ++i) { v[i] = x[tid + i * 256]; s += v[i]; }
    red[tid] = s; __syncthreads();
    for (int off = 128; off > 0; off >>= 1) {
        if (tid < off) red[tid] += red[tid + off];
        __syncthreads();
    }
    float mu = red[0] * (1.0f / Dd);
    __syncthreads();
    float s2 = 0.f;
#pragma unroll
    for (int i = 0; i < 4; ++i) { float d0 = v[i] - mu; s2 += d0 * d0; }
    red[tid] = s2; __syncthreads();
    for (int off = 128; off > 0; off >>= 1) {
        if (tid < off) red[tid] += red[tid + off];
        __syncthreads();
    }
    float rs = rsqrtf(red[0] * (1.0f / Dd) + 1e-5f);
    bf16_t* o = out + (size_t)blockIdx.x * Dd;
#pragma unroll
    for (int i = 0; i < 4; ++i) {
        int d = tid + i * 256;
        o[d] = (bf16_t)((v[i] - mu) * rs * g[d] + b[d]);
    }
}

// ---------------------------------------------------------------------------
// RoPE in place on the Q and K slices of qkv [Mtok, 3*D] (f32).
// ---------------------------------------------------------------------------
__global__ __launch_bounds__(256) void rope_kernel(float* __restrict__ qkv)
{
    int idx = blockIdx.x * 256 + threadIdx.x;   // Mtok*Hh*2*32 threads
    int j   = idx & 31;
    int t   = idx >> 5;
    int p   = t & 1;            // 0 = q, 1 = k
    t >>= 1;
    int hh  = t & 15;
    int m   = t >> 4;
    int pos = m & (Ss - 1);
    float* base = qkv + (size_t)m * (3 * Dd) + p * Dd + hh * DH;
    float x1 = base[j];
    float x2 = base[j + 32];
    float freq = __expf((float)j * (-2.0f / 64.0f) * 9.2103403719761836f);
    float ang  = (float)pos * freq;
    float sn = sinf(ang), cs = cosf(ang);
    base[j]      = x1 * cs - x2 * sn;
    base[j + 32] = x2 * cs + x1 * sn;
}

// ---------------------------------------------------------------------------
// GEMM: C[M,N] = epilogue( A[M,K](bf16) @ B[N,K](bf16)^T (+ Res) )
// Double-buffered TDM pipeline: wave 0 issues next K-panel's DMA before
// computing the current one, then s_wait_tensorcnt(2) — the 2 in-flight ops
// are the next tile; current tile (issued earlier, in-order) is complete.
// Tiles 128x128x32, 256 threads = 8 waves, wave computes 64x32 (4x2 WMMA).
// EPI: 0 = f32 store, 1 = f32 Res+acc, 2 = exact GELU -> bf16 store
// ---------------------------------------------------------------------------
template <int EPI>
__global__ __launch_bounds__(256) void gemm_bf16_kernel(
    void* __restrict__ Cv, const bf16_t* __restrict__ A,
    const bf16_t* __restrict__ B, const float* __restrict__ Res,
    int M, int N, int K)
{
    __shared__ __align__(16) bf16_t As[2][128][32];
    __shared__ __align__(16) bf16_t Bs[2][128][32];

    const int tid  = threadIdx.x;
    const int lane = tid & 31;
    const int wid  = tid >> 5;
    const int wr   = wid & 1;
    const int wc   = wid >> 1;
    const int l15  = lane & 15;
    const int lhi  = lane >> 4;
    const int bm   = blockIdx.y * 128;
    const int bn   = blockIdx.x * 128;
    const int nk   = K >> 5;                 // K / 32 panels

    v8f acc[4][2];
#pragma unroll
    for (int mt = 0; mt < 4; ++mt)
#pragma unroll
        for (int nt = 0; nt < 2; ++nt) acc[mt][nt] = zero8();

    if (wid == 0) {                          // prologue DMA: panel 0 -> buf 0
        tdm_load_2d(A + (size_t)bm * K, (u32)(uintptr_t)&As[0][0][0],
                    32u, 128u, (u32)K);
        tdm_load_2d(B + (size_t)bn * K, (u32)(uintptr_t)&Bs[0][0][0],
                    32u, 128u, (u32)K);
    }

    for (int i = 0; i < nk; ++i) {
        const int cur = i & 1;
        if (wid == 0) {
            if (i + 1 < nk) {                // DMA next panel into other buffer
                const int nxt = cur ^ 1;
                const int kk  = (i + 1) << 5;
                tdm_load_2d(A + (size_t)bm * K + kk,
                            (u32)(uintptr_t)&As[nxt][0][0], 32u, 128u, (u32)K);
                tdm_load_2d(B + (size_t)bn * K + kk,
                            (u32)(uintptr_t)&Bs[nxt][0][0], 32u, 128u, (u32)K);
                __builtin_amdgcn_s_wait_tensorcnt((short)2);  // current landed
            } else {
                __builtin_amdgcn_s_wait_tensorcnt((short)0);
            }
        }
        if (wid == 1 && i + 2 < nk) {        // L2 hint two panels ahead
            const int kk = (i + 2) << 5;
            __builtin_prefetch(A + (size_t)(bm + lane * 4) * K + kk, 0, 1);
            __builtin_prefetch(B + (size_t)(bn + lane * 4) * K + kk, 0, 1);
        }
        __syncthreads();

        FragBF af[4], bf[2];
#pragma unroll
        for (int mt = 0; mt < 4; ++mt) {
            int row = wr * 64 + mt * 16 + l15;
            af[mt].h[0] = *(const v8bf*)&As[cur][row][lhi * 8];
            af[mt].h[1] = *(const v8bf*)&As[cur][row][lhi * 8 + 16];
        }
#pragma unroll
        for (int nt = 0; nt < 2; ++nt) {
            int col = wc * 32 + nt * 16 + l15;
            bf[nt].h[0] = *(const v8bf*)&Bs[cur][col][lhi * 8];
            bf[nt].h[1] = *(const v8bf*)&Bs[cur][col][lhi * 8 + 16];
        }
#pragma unroll
        for (int mt = 0; mt < 4; ++mt)
#pragma unroll
            for (int nt = 0; nt < 2; ++nt)
                acc[mt][nt] = wmma_bf16(af[mt].v, bf[nt].v, acc[mt][nt]);
        __syncthreads();   // drain buf[cur] before it is DMA-overwritten
    }

    // C layout: VGPR j -> M=j (lanes 0-15) / M=j+8 (lanes 16-31); N = lane&15
#pragma unroll
    for (int mt = 0; mt < 4; ++mt) {
#pragma unroll
        for (int nt = 0; nt < 2; ++nt) {
#pragma unroll
            for (int j = 0; j < 8; ++j) {
                int row = bm + wr * 64 + mt * 16 + j + (lhi << 3);
                int col = bn + wc * 32 + nt * 16 + l15;
                size_t idx = (size_t)row * N + col;
                float v = acc[mt][nt][j];
                if (EPI == 0) {
                    ((float*)Cv)[idx] = v;
                } else if (EPI == 1) {
                    ((float*)Cv)[idx] = v + Res[idx];
                } else {
                    v = 0.5f * v * (1.0f + erff(v * 0.70710678118654752f));
                    ((bf16_t*)Cv)[idx] = (bf16_t)v;
                }
            }
        }
    }
}

// ---------------------------------------------------------------------------
// Flash-style causal attention. Grid: (S/64, B*H). Block: 256 threads.
// qkv f32 [Mtok, 3*D]; output bf16 [Mtok, D] (feeds proj GEMM A).
// ---------------------------------------------------------------------------
__global__ __launch_bounds__(256) void attn_kernel(
    bf16_t* __restrict__ O, const float* __restrict__ QKV)
{
    __shared__ __align__(16) bf16_t Qs[64][64];
    __shared__ __align__(16) bf16_t Ks[64][64];
    __shared__ __align__(16) bf16_t Vt[64][64];   // transposed: [dh][key]
    __shared__ __align__(16) float  Ssc[64][64];
    __shared__ __align__(16) bf16_t Ps[64][64];
    __shared__ float rowM[64], rowL[64], rowScale[64];

    const int tid  = threadIdx.x;
    const int lane = tid & 31;
    const int wid  = tid >> 5;
    const int wr   = wid & 3;
    const int wc   = wid >> 2;
    const int l15  = lane & 15;
    const int lhi  = lane >> 4;

    const int qb = blockIdx.x;
    const int bh = blockIdx.y;
    const int b  = bh >> 4;
    const int hh = bh & 15;
    const int q0 = qb * 64;

    const float* qbase = QKV + (size_t)(b * Ss + q0) * (3 * Dd) + hh * DH;
#pragma unroll
    for (int u = tid, it = 0; it < 4; ++it, u += 256) {
        int r = u >> 4;
        int c = (u & 15) << 2;
        float4 f = *(const float4*)(qbase + (size_t)r * (3 * Dd) + c);
        Qs[r][c + 0] = (bf16_t)f.x; Qs[r][c + 1] = (bf16_t)f.y;
        Qs[r][c + 2] = (bf16_t)f.z; Qs[r][c + 3] = (bf16_t)f.w;
    }
    if (tid < 64) { rowM[tid] = -3.0e38f; rowL[tid] = 0.0f; }
    __syncthreads();

    FragBF qf[2];
#pragma unroll
    for (int kc = 0; kc < 2; ++kc) {
        int row = wr * 16 + l15;
        qf[kc].h[0] = *(const v8bf*)&Qs[row][kc * 32 + lhi * 8];
        qf[kc].h[1] = *(const v8bf*)&Qs[row][kc * 32 + lhi * 8 + 16];
    }

    v8f oacc[2];
#pragma unroll
    for (int nt = 0; nt < 2; ++nt) oacc[nt] = zero8();

    for (int kt = 0; kt <= qb; ++kt) {
        const float* kbase = QKV + (size_t)(b * Ss + kt * 64) * (3 * Dd) + Dd + hh * DH;
        const float* vbase = kbase + Dd;
#pragma unroll
        for (int u = tid, it = 0; it < 4; ++it, u += 256) {
            int r = u >> 4;
            int c = (u & 15) << 2;
            float4 fk = *(const float4*)(kbase + (size_t)r * (3 * Dd) + c);
            Ks[r][c + 0] = (bf16_t)fk.x; Ks[r][c + 1] = (bf16_t)fk.y;
            Ks[r][c + 2] = (bf16_t)fk.z; Ks[r][c + 3] = (bf16_t)fk.w;
            float4 fv = *(const float4*)(vbase + (size_t)r * (3 * Dd) + c);
            Vt[c + 0][r] = (bf16_t)fv.x; Vt[c + 1][r] = (bf16_t)fv.y;
            Vt[c + 2][r] = (bf16_t)fv.z; Vt[c + 3][r] = (bf16_t)fv.w;
        }
        __syncthreads();

        // S = Q @ K^T / sqrt(DH)
#pragma unroll
        for (int nt = 0; nt < 2; ++nt) {
            FragBF kf0, kf1;
            int key = wc * 32 + nt * 16 + l15;
            kf0.h[0] = *(const v8bf*)&Ks[key][lhi * 8];
            kf0.h[1] = *(const v8bf*)&Ks[key][lhi * 8 + 16];
            kf1.h[0] = *(const v8bf*)&Ks[key][32 + lhi * 8];
            kf1.h[1] = *(const v8bf*)&Ks[key][32 + lhi * 8 + 16];
            v8f sf = zero8();
            sf = wmma_bf16(qf[0].v, kf0.v, sf);
            sf = wmma_bf16(qf[1].v, kf1.v, sf);
#pragma unroll
            for (int j = 0; j < 8; ++j) {
                int row = wr * 16 + j + (lhi << 3);
                int col = wc * 32 + nt * 16 + l15;
                Ssc[row][col] = sf[j] * 0.125f;
            }
        }
        __syncthreads();

        // Online softmax, one thread per query row
        if (tid < 64) {
            int r = tid;
            int ncols = (kt == qb) ? (r + 1) : 64;
            float mold = rowM[r];
            float mnew = mold;
            for (int c = 0; c < ncols; ++c) mnew = fmaxf(mnew, Ssc[r][c]);
            float sc = __expf(mold - mnew);
            float l  = rowL[r] * sc;
            for (int c = 0; c < 64; ++c) {
                float e = (c < ncols) ? __expf(Ssc[r][c] - mnew) : 0.0f;
                Ps[r][c] = (bf16_t)e;
                l += e;
            }
            rowM[r] = mnew; rowL[r] = l; rowScale[r] = sc;
        }
        __syncthreads();

        // O = O*scale + P @ V
        FragBF pf[2];
#pragma unroll
        for (int kc = 0; kc < 2; ++kc) {
            int row = wr * 16 + l15;
            pf[kc].h[0] = *(const v8bf*)&Ps[row][kc * 32 + lhi * 8];
            pf[kc].h[1] = *(const v8bf*)&Ps[row][kc * 32 + lhi * 8 + 16];
        }
#pragma unroll
        for (int nt = 0; nt < 2; ++nt) {
            int dcol = wc * 32 + nt * 16 + l15;
            FragBF vf0, vf1;
            vf0.h[0] = *(const v8bf*)&Vt[dcol][lhi * 8];
            vf0.h[1] = *(const v8bf*)&Vt[dcol][lhi * 8 + 16];
            vf1.h[0] = *(const v8bf*)&Vt[dcol][32 + lhi * 8];
            vf1.h[1] = *(const v8bf*)&Vt[dcol][32 + lhi * 8 + 16];
#pragma unroll
            for (int j = 0; j < 8; ++j)
                oacc[nt][j] *= rowScale[wr * 16 + j + (lhi << 3)];
            oacc[nt] = wmma_bf16(pf[0].v, vf0.v, oacc[nt]);
            oacc[nt] = wmma_bf16(pf[1].v, vf1.v, oacc[nt]);
        }
        __syncthreads();
    }

#pragma unroll
    for (int nt = 0; nt < 2; ++nt) {
#pragma unroll
        for (int j = 0; j < 8; ++j) {
            int row = wr * 16 + j + (lhi << 3);
            int col = wc * 32 + nt * 16 + l15;
            float invl = 1.0f / rowL[row];
            O[(size_t)(b * Ss + q0 + row) * Dd + hh * DH + col] =
                (bf16_t)(oacc[nt][j] * invl);
        }
    }
}

// ---------------------------------------------------------------------------
// Host-side orchestration
// ---------------------------------------------------------------------------
extern "C" void kernel_launch(void* const* d_in, const int* in_sizes, int n_in,
                              void* d_out, int out_size, void* d_ws, size_t ws_size,
                              hipStream_t stream) {
    const int*   ids  = (const int*)d_in[0];
    const float* tok  = (const float*)d_in[1];
    const float* Wqkv = (const float*)d_in[2];
    const float* Wprj = (const float*)d_in[3];
    const float* Wfc1 = (const float*)d_in[4];
    const float* Wfc2 = (const float*)d_in[5];
    const float* g1   = (const float*)d_in[6];
    const float* b1   = (const float*)d_in[7];
    const float* g2   = (const float*)d_in[8];
    const float* b2   = (const float*)d_in[9];
    const float* gf   = (const float*)d_in[10];
    const float* bfb  = (const float*)d_in[11];
    float* out = (float*)d_out;

    // Workspace: f32 residual/qkv + bf16 activations + rotating bf16 weight buf
    float*  x    = (float*)d_ws;                           // [2048,1024] f32
    float*  qkv  = x + (size_t)Mtok * Dd;                  // [2048,3072] f32
    bf16_t* hb   = (bf16_t*)(qkv + (size_t)Mtok * 3 * Dd); // [2048,1024] bf16
    bf16_t* ob   = hb + (size_t)Mtok * Dd;                 // [2048,1024] bf16
    bf16_t* ffnb = ob + (size_t)Mtok * Dd;                 // [2048,4096] bf16
    bf16_t* wbuf = ffnb + (size_t)Mtok * INNER;            // [32000,1024] bf16 max

    embed_kernel<<<(Mtok * Dd) / 256, 256, 0, stream>>>(x, tok, ids);

    for (int l = 0; l < Ll; ++l) {
        // qkv = LN1(x) @ Wqkv^T
        cvt_bf16_kernel<<<(3 * Dd * Dd) / 1024, 256, 0, stream>>>(
            wbuf, Wqkv + (size_t)l * 3 * Dd * Dd);
        layernorm_kernel<<<Mtok, 256, 0, stream>>>(hb, x, g1 + l * Dd, b1 + l * Dd);
        gemm_bf16_kernel<0><<<dim3(3 * Dd / 128, Mtok / 128), 256, 0, stream>>>(
            qkv, hb, wbuf, nullptr, Mtok, 3 * Dd, Dd);
        rope_kernel<<<(Mtok * Hh * 2 * 32) / 256, 256, 0, stream>>>(qkv);
        attn_kernel<<<dim3(Ss / 64, Bb * Hh), 256, 0, stream>>>(ob, qkv);
        // x = x + o @ Wproj^T
        cvt_bf16_kernel<<<(Dd * Dd) / 1024, 256, 0, stream>>>(
            wbuf, Wprj + (size_t)l * Dd * Dd);
        gemm_bf16_kernel<1><<<dim3(Dd / 128, Mtok / 128), 256, 0, stream>>>(
            x, ob, wbuf, x, Mtok, Dd, Dd);
        // ffn = GELU(LN2(x) @ Wfc1^T)
        layernorm_kernel<<<Mtok, 256, 0, stream>>>(hb, x, g2 + l * Dd, b2 + l * Dd);
        cvt_bf16_kernel<<<(INNER * Dd) / 1024, 256, 0, stream>>>(
            wbuf, Wfc1 + (size_t)l * INNER * Dd);
        gemm_bf16_kernel<2><<<dim3(INNER / 128, Mtok / 128), 256, 0, stream>>>(
            ffnb, hb, wbuf, nullptr, Mtok, INNER, Dd);
        // x = x + ffn @ Wfc2^T
        cvt_bf16_kernel<<<(Dd * INNER) / 1024, 256, 0, stream>>>(
            wbuf, Wfc2 + (size_t)l * Dd * INNER);
        gemm_bf16_kernel<1><<<dim3(Dd / 128, Mtok / 128), 256, 0, stream>>>(
            x, ffnb, wbuf, x, Mtok, Dd, INNER);
    }

    // logits = LNf(x) @ tok_emb^T
    layernorm_kernel<<<Mtok, 256, 0, stream>>>(hb, x, gf, bfb);
    cvt_bf16_kernel<<<(Vv * Dd) / 1024, 256, 0, stream>>>(wbuf, tok);
    gemm_bf16_kernel<0><<<dim3(Vv / 128, Mtok / 128), 256, 0, stream>>>(
        out, hb, wbuf, nullptr, Mtok, Vv, Dd);
}
